// Block_86835648790988
// MI455X (gfx1250) — matrix operand order repaired and checked by reading, not compiled, not run on hardware
//
#include <hip/hip_runtime.h>
#include <cstdint>
#include <cstddef>

typedef __attribute__((ext_vector_type(16))) _Float16 v16h;
typedef __attribute__((ext_vector_type(8)))  _Float16 v8h;
typedef __attribute__((ext_vector_type(8)))  float    v8f;

#define N_EMBED 384
#define N_HEADS 6
#define SEQT    256
#define BATB    128
#define BT      (BATB * SEQT)      /* 32768 rows */
#define LN_EPS  1e-5f
#define ATT_SCALE 0.05103103630798287f  /* 384^-0.5 (reference scales by C, not head dim) */

// ---------------------------------------------------------------------------
// Tensor Data Mover (TDM) tile load: 2D tile -> LDS with padding, per ISA ch.8
// ---------------------------------------------------------------------------
#if defined(__gfx1250__) && __has_builtin(__builtin_amdgcn_tensor_load_to_lds) && \
    __has_builtin(__builtin_amdgcn_s_wait_tensorcnt)
#define USE_TDM 1
typedef unsigned int u32x4 __attribute__((ext_vector_type(4)));
typedef int          i32x8 __attribute__((ext_vector_type(8)));
typedef int          i32x4 __attribute__((ext_vector_type(4)));

// Load a 128-row x 32-half tile from a row-major f16 tensor (row stride = K
// halves) into LDS at `lds_dst`, padding 4 DWORDs after every 16 DWORDs
// (one 32-half row) -> LDS row stride 40 halves, matching frag loaders.
static __device__ __forceinline__ void tdm_load_tile_a(const _Float16* gsrc,
                                                       void* lds_dst, int K) {
  u32x4 g0;
  g0.x = 1u;                                    // count=1 (valid user descriptor)
  g0.y = (unsigned int)(uintptr_t)lds_dst;      // lds_addr (bytes)
  const unsigned long long ga = (unsigned long long)(uintptr_t)gsrc;
  g0.z = (unsigned int)(ga & 0xffffffffu);      // global_addr[31:0]
  g0.w = (unsigned int)((ga >> 32) & 0x01ffffffu) | 0x80000000u;  // addr[56:32] | type=2

  const unsigned int td0 = (unsigned int)K;     // tensor_dim0 (elements)
  const unsigned int td1 = 32768u;              // tensor_dim1 (rows)
  i32x8 g1;
  // data_size=1 (2B), pad_enable, pad_interval=3 (16 DWORDs), pad_amount=3 (4 DWORDs)
  g1[0] = (1 << 16) | (1 << 20) | (3 << 22) | (3 << 25);
  g1[1] = (int)((td0 & 0xffffu) << 16);                       // tensor_dim0[15:0]
  g1[2] = (int)((td0 >> 16) | ((td1 & 0xffffu) << 16));       // dim0 hi | dim1 lo
  g1[3] = (int)((td1 >> 16) | (32u << 16));                   // dim1 hi | tile_dim0=32
  g1[4] = (int)128u;                                          // tile_dim1=128, tile_dim2=0
  g1[5] = (int)(unsigned int)K;                               // tensor_dim0_stride lo32
  g1[6] = 0;                                                  // stride hi16 | dim1_stride lo
  g1[7] = 0;
  const i32x4 gz4 = {0, 0, 0, 0};
  const i32x8 gz8 = {0, 0, 0, 0, 0, 0, 0, 0};
  // 6-arg toolchain variant: (g0, g1, g2, g3, <i32x8>, cpol)
  __builtin_amdgcn_tensor_load_to_lds(g0, g1, gz4, gz4, gz8, 0);
}
#endif

// ---------------------------------------------------------------------------
// WMMA fragment helpers (layouts per CDNA5 ISA 7.12.2)
// ---------------------------------------------------------------------------

// Two adjacent v8h registers ARE a v16h operand: pure register placement.
static __device__ __forceinline__ v16h join16(v8h lo, v8h hi) {
  union { v8h p[2]; v16h v; } u;
  u.p[0] = lo; u.p[1] = hi;
  return u.v;
}

// A-matrix 16x32 f16 from row-major [M][K] tile, leading dim `ld` (halves).
// lane&15 = row M; lanes<16 hold K = {0..7, 16..23}, lanes>=16 K = {8..15, 24..31}.
static __device__ __forceinline__ v16h frag_a(const _Float16* base, int ld, int lane) {
  const int row = lane & 15, hi = lane >> 4;
  const v8h x0 = *(const v8h*)(base + row * ld + hi * 8);
  const v8h x1 = *(const v8h*)(base + row * ld + 16 + hi * 8);
  return join16(x0, x1);
}

// B-matrix 32x16 f16 from K-contiguous [N][K] tile, leading dim `ld` (halves).
// lane&15 = column N; lanes<16 hold K = {0..7, 16..23}, lanes>=16 K = {8..15, 24..31}.
static __device__ __forceinline__ v16h frag_b_t(const _Float16* base, int ld, int lane) {
  const int col = lane & 15, hi = lane >> 4;
  const v8h y0 = *(const v8h*)(base + col * ld + hi * 8);
  const v8h y1 = *(const v8h*)(base + col * ld + 16 + hi * 8);
  return join16(y0, y1);
}

#define WMMA_F16(A, B, C) \
  __builtin_amdgcn_wmma_f32_16x16x32_f16(false, (A), false, (B), (short)0, (C), false, false)

// ---------------------------------------------------------------------------
// Weight conversion kernels (f32 -> f16, QKV fused into one [384,1152] matrix)
// ---------------------------------------------------------------------------

__global__ __launch_bounds__(256) void k_conv_f16(const float* __restrict__ src,
                                                  _Float16* __restrict__ dst, int n) {
  int i = blockIdx.x * 256 + threadIdx.x;
  if (i < n) dst[i] = (_Float16)src[i];
}

// dst[c][j], j<384:q, <768:k, else v; within slice col = h*64+d, src W[h][c][d]
__global__ __launch_bounds__(256) void k_pack_qkv(const float* __restrict__ Wq,
                                                  const float* __restrict__ Wk,
                                                  const float* __restrict__ Wv,
                                                  _Float16* __restrict__ dst) {
  int i = blockIdx.x * 256 + threadIdx.x;
  if (i >= 384 * 1152) return;
  int c = i / 1152, j = i % 1152;
  int w = j / 384, jj = j % 384;
  int h = jj >> 6, d = jj & 63;
  const float* W = (w == 0) ? Wq : (w == 1) ? Wk : Wv;
  dst[i] = (_Float16)W[((h * 384 + c) << 6) + d];
}

// ---------------------------------------------------------------------------
// LayerNorm: one 128-thread block per row of 384, f32 in -> f16 out
// ---------------------------------------------------------------------------

__global__ __launch_bounds__(128) void k_layernorm(const float* __restrict__ x,
                                                   const float* __restrict__ g,
                                                   const float* __restrict__ b,
                                                   _Float16* __restrict__ out) {
  __shared__ float red[128];
  const int t = threadIdx.x;
  const size_t base = (size_t)blockIdx.x * N_EMBED;
  float v0 = x[base + t], v1 = x[base + t + 128], v2 = x[base + t + 256];
  red[t] = v0 + v1 + v2;
  __syncthreads();
  for (int o = 64; o > 0; o >>= 1) { if (t < o) red[t] += red[t + o]; __syncthreads(); }
  float mu = red[0] * (1.0f / N_EMBED);
  __syncthreads();
  float d0 = v0 - mu, d1 = v1 - mu, d2 = v2 - mu;
  red[t] = d0 * d0 + d1 * d1 + d2 * d2;
  __syncthreads();
  for (int o = 64; o > 0; o >>= 1) { if (t < o) red[t] += red[t + o]; __syncthreads(); }
  float rs = rsqrtf(red[0] * (1.0f / N_EMBED) + LN_EPS);
  out[base + t]       = (_Float16)(d0 * rs * g[t]       + b[t]);
  out[base + t + 128] = (_Float16)(d1 * rs * g[t + 128] + b[t + 128]);
  out[base + t + 256] = (_Float16)(d2 * rs * g[t + 256] + b[t + 256]);
}

// ---------------------------------------------------------------------------
// Pipelined WMMA GEMM: C[M,N] = A[M,K](f16) * B[K,N](f16), f32 accumulate.
// Block tile 128x128, 8 waves (4x2), 32x64 per wave, BK=32, double-buffered
// LDS, one barrier per K-step. A tiles via TDM DMA (TENSORcnt) when
// available; B staged K-contiguous ([N][K]) so fragments are ds_load_b128.
// EPI: 0 = store f16 | 1 = f32: acc + bias[col] + res[idx] | 2 = f16 relu(acc+bias)
// ---------------------------------------------------------------------------

template <int EPI>
__global__ __launch_bounds__(256) void k_gemm(const _Float16* __restrict__ A,
                                              const _Float16* __restrict__ B,
                                              int K, int N,
                                              const float* __restrict__ bias,
                                              const float* __restrict__ res,
                                              void* __restrict__ outv) {
  __shared__ _Float16 As[2][128 * 40];  // [M][K], row stride 40 halves (80B, 16B-aligned)
  __shared__ _Float16 Bt[2][128 * 40];  // [N][K], row stride 40 halves
  const int tid = threadIdx.x;
  const int lane = tid & 31, wid = tid >> 5;
  const int m0 = blockIdx.y * 128, n0 = blockIdx.x * 128;
  const int wm = (wid >> 1) * 32, wn = (wid & 1) * 64;
  const int bkr = tid >> 3, bnc = (tid & 7) * 16;   // B: row k = tid/8, 16 n-values each
#ifndef USE_TDM
  const int ar = tid >> 1, aseg = (tid & 1) * 16;   // A: 2 threads/row, 16 halves each
#endif
  const v8f vz = {};

  v8f acc[2][4];
#pragma unroll
  for (int mt = 0; mt < 2; ++mt)
#pragma unroll
    for (int nt = 0; nt < 4; ++nt) acc[mt][nt] = vz;

  const int nK = K >> 5;
  v8h pb0, pb1;
#ifndef USE_TDM
  v8h pa0, pa1;
#endif

  // prologue: fetch tile 0
  {
#ifdef USE_TDM
    if (wid == 0) tdm_load_tile_a(A + (size_t)m0 * K, &As[0][0], K);
#else
    const _Float16* gA = A + (size_t)(m0 + ar) * K + aseg;
    pa0 = *(const v8h*)(gA);
    pa1 = *(const v8h*)(gA + 8);
    *(v8h*)(&As[0][ar * 40 + aseg])     = pa0;
    *(v8h*)(&As[0][ar * 40 + aseg + 8]) = pa1;
#endif
    const _Float16* gB = B + (size_t)bkr * N + n0 + bnc;
    pb0 = *(const v8h*)(gB);
    pb1 = *(const v8h*)(gB + 8);
#pragma unroll
    for (int i = 0; i < 8; ++i) {
      Bt[0][(bnc + i) * 40 + bkr]     = pb0[i];
      Bt[0][(bnc + 8 + i) * 40 + bkr] = pb1[i];
    }
#ifdef USE_TDM
    if (wid == 0) __builtin_amdgcn_s_wait_tensorcnt(0);
#endif
  }
  __syncthreads();

  for (int t = 0; t < nK; ++t) {
    const int cur = t & 1;
    const bool more = (t + 1) < nK;
    if (more) {  // prefetch next tile (DMA / registers) overlapping WMMA below
      const int kt = (t + 1) << 5;
#ifdef USE_TDM
      if (wid == 0) tdm_load_tile_a(A + (size_t)m0 * K + kt, &As[cur ^ 1][0], K);
#else
      const _Float16* gA = A + (size_t)(m0 + ar) * K + kt + aseg;
      pa0 = *(const v8h*)(gA);
      pa1 = *(const v8h*)(gA + 8);
#endif
      const _Float16* gB = B + (size_t)(kt + bkr) * N + n0 + bnc;
      pb0 = *(const v8h*)(gB);
      pb1 = *(const v8h*)(gB + 8);
    }

    const v16h a0 = frag_a(&As[cur][wm * 40], 40, lane);
    const v16h a1 = frag_a(&As[cur][(wm + 16) * 40], 40, lane);
#pragma unroll
    for (int nt = 0; nt < 4; ++nt) {
      const v16h bf = frag_b_t(&Bt[cur][(wn + nt * 16) * 40], 40, lane);
      acc[0][nt] = WMMA_F16(a0, bf, acc[0][nt]);
      acc[1][nt] = WMMA_F16(a1, bf, acc[1][nt]);
    }

    if (more) {
      const int nxt = cur ^ 1;
#ifndef USE_TDM
      *(v8h*)(&As[nxt][ar * 40 + aseg])     = pa0;
      *(v8h*)(&As[nxt][ar * 40 + aseg + 8]) = pa1;
#endif
#pragma unroll
      for (int i = 0; i < 8; ++i) {
        Bt[nxt][(bnc + i) * 40 + bkr]     = pb0[i];
        Bt[nxt][(bnc + 8 + i) * 40 + bkr] = pb1[i];
      }
#ifdef USE_TDM
      if (wid == 0) __builtin_amdgcn_s_wait_tensorcnt(0);
#endif
    }
    __syncthreads();
  }

  const int hi = lane >> 4, cl = lane & 15;
#pragma unroll
  for (int mt = 0; mt < 2; ++mt) {
#pragma unroll
    for (int nt = 0; nt < 4; ++nt) {
      const int col = n0 + wn + nt * 16 + cl;
#pragma unroll
      for (int r = 0; r < 8; ++r) {
        const int row = m0 + wm + mt * 16 + hi * 8 + r;
        const size_t idx = (size_t)row * N + col;
        const float v = acc[mt][nt][r];
        if (EPI == 0)      ((_Float16*)outv)[idx] = (_Float16)v;
        else if (EPI == 1) ((float*)outv)[idx]    = v + bias[col] + res[idx];
        else               ((_Float16*)outv)[idx] = (_Float16)fmaxf(v + bias[col], 0.f);
      }
    }
  }
}

// ---------------------------------------------------------------------------
// Causal flash attention: one block per (b,h); 8 waves x 32 query rows.
// qkv layout: [b*T + t][1152] = q(h*64+d) | k(+384) | v(+768), f16.
// K tile staged [key][d] (score B-frags contiguous along d); V tile staged
// transposed [d][key] (PV B-frags contiguous along key). Online softmax in
// the C/D register layout; P re-enters A layout via per-wave LDS patch.
// ---------------------------------------------------------------------------

__global__ __launch_bounds__(256) void k_attention(const _Float16* __restrict__ qkv,
                                                   _Float16* __restrict__ att) {
  __shared__ _Float16 ks[32 * 72];       // [key][d], stride 72 (144B, 16B-aligned)
  __shared__ _Float16 vt[64 * 40];       // [d][key], stride 40 (80B, 16B-aligned)
  __shared__ _Float16 Ps[8 * 32 * 40];   // per-wave P scratch, [32 rows][32 keys]
  const int tid = threadIdx.x, lane = tid & 31, wid = tid >> 5;
  const int bIdx = blockIdx.x / N_HEADS, h = blockIdx.x % N_HEADS;
  const int hi = lane >> 4, cl = lane & 15;
  const int q0 = wid * 32;
  const size_t rowstride = 1152;
  const size_t qbase = (size_t)(bIdx * SEQT) * rowstride + h * 64;
  const v8f vz = {};

  // Q fragments (A layout) straight from global: 2 M-tiles x 2 K-chunks of d
  v16h aq[2][2];
#pragma unroll
  for (int mt = 0; mt < 2; ++mt) {
    const _Float16* qp = qkv + qbase + (size_t)(q0 + mt * 16 + cl) * rowstride;
#pragma unroll
    for (int kc = 0; kc < 2; ++kc) {
      const v8h x0 = *(const v8h*)(qp + kc * 32 + hi * 8);
      const v8h x1 = *(const v8h*)(qp + kc * 32 + 16 + hi * 8);
      aq[mt][kc] = join16(x0, x1);
    }
  }

  v8f O[2][4];
  float m[2][8], l[2][8];
#pragma unroll
  for (int mt = 0; mt < 2; ++mt) {
#pragma unroll
    for (int dt = 0; dt < 4; ++dt) O[mt][dt] = vz;
#pragma unroll
    for (int r = 0; r < 8; ++r) { m[mt][r] = -1e30f; l[mt][r] = 0.f; }
  }

  const int krow = tid >> 3, kcol8 = (tid & 7) * 8;
  _Float16* Pw = Ps + wid * 32 * 40;

  for (int kb = 0; kb < SEQT / 32; ++kb) {
    __syncthreads();
    {
      const size_t src = (size_t)(bIdx * SEQT + kb * 32 + krow) * rowstride + h * 64 + kcol8;
      *(v8h*)(ks + krow * 72 + kcol8) = *(const v8h*)(qkv + src + 384);
      const v8h pv = *(const v8h*)(qkv + src + 768);
#pragma unroll
      for (int i = 0; i < 8; ++i) vt[(kcol8 + i) * 40 + krow] = pv[i];  // transpose V
    }
    __syncthreads();
    if (kb > wid) continue;  // causal: this wave's rows never see later key blocks

    // S = Q * K^T  (B fragment is k^T: contiguous along d per lane)
    v8f s[2][2] = {{vz, vz}, {vz, vz}};
#pragma unroll
    for (int kc = 0; kc < 2; ++kc) {
#pragma unroll
      for (int nt = 0; nt < 2; ++nt) {
        const _Float16* kp = ks + (nt * 16 + cl) * 72 + kc * 32 + hi * 8;
        const v16h bk = join16(*(const v8h*)(kp), *(const v8h*)(kp + 16));
        s[0][nt] = WMMA_F16(aq[0][kc], bk, s[0][nt]);
        s[1][nt] = WMMA_F16(aq[1][kc], bk, s[1][nt]);
      }
    }

    // scale + causal mask + online softmax
#pragma unroll
    for (int mt = 0; mt < 2; ++mt) {
      const int rowg0 = q0 + mt * 16 + hi * 8;
      float rmax[8];
#pragma unroll
      for (int r = 0; r < 8; ++r) {
#pragma unroll
        for (int nt = 0; nt < 2; ++nt) {
          const int kg = kb * 32 + nt * 16 + cl;
          float sv = s[mt][nt][r] * ATT_SCALE;
          if (kg > rowg0 + r) sv = -1e30f;
          s[mt][nt][r] = sv;
        }
        float rm = fmaxf(s[mt][0][r], s[mt][1][r]);
        rm = fmaxf(rm, __shfl_xor(rm, 1, 16));
        rm = fmaxf(rm, __shfl_xor(rm, 2, 16));
        rm = fmaxf(rm, __shfl_xor(rm, 4, 16));
        rm = fmaxf(rm, __shfl_xor(rm, 8, 16));
        rmax[r] = rm;
      }
#pragma unroll
      for (int r = 0; r < 8; ++r) {
        const float mn = fmaxf(m[mt][r], rmax[r]);
        const float corr = __expf(m[mt][r] - mn);
        m[mt][r] = mn;
        const float p0 = __expf(s[mt][0][r] - mn);
        const float p1 = __expf(s[mt][1][r] - mn);
        float rs = p0 + p1;
        rs += __shfl_xor(rs, 1, 16);
        rs += __shfl_xor(rs, 2, 16);
        rs += __shfl_xor(rs, 4, 16);
        rs += __shfl_xor(rs, 8, 16);
        l[mt][r] = l[mt][r] * corr + rs;
#pragma unroll
        for (int dt = 0; dt < 4; ++dt) O[mt][dt][r] *= corr;
        const int rl = mt * 16 + hi * 8 + r;   // stash P (per-wave LDS, in-order DS)
        Pw[rl * 40 + cl]      = (_Float16)p0;
        Pw[rl * 40 + 16 + cl] = (_Float16)p1;
      }
    }

    // O += P * V  (V^T staged in LDS: fragment = two ds_load_b128)
    const v16h ap0 = frag_a(Pw, 40, lane);
    const v16h ap1 = frag_a(Pw + 16 * 40, 40, lane);
#pragma unroll
    for (int dt = 0; dt < 4; ++dt) {
      const v16h bv = frag_b_t(vt + (dt * 16) * 40, 40, lane);
      O[0][dt] = WMMA_F16(ap0, bv, O[0][dt]);
      O[1][dt] = WMMA_F16(ap1, bv, O[1][dt]);
    }
  }

  // normalize + write att[b*T + t][h*64 + d] (head-concat feature layout)
#pragma unroll
  for (int mt = 0; mt < 2; ++mt) {
#pragma unroll
    for (int r = 0; r < 8; ++r) {
      const int row = q0 + mt * 16 + hi * 8 + r;
      const float inv = 1.0f / l[mt][r];
      const size_t ob = (size_t)(bIdx * SEQT + row) * N_EMBED + h * 64;
#pragma unroll
      for (int dt = 0; dt < 4; ++dt)
        att[ob + dt * 16 + cl] = (_Float16)(O[mt][dt][r] * inv);
    }
  }
}

// ---------------------------------------------------------------------------
// Launcher
// ---------------------------------------------------------------------------

extern "C" void kernel_launch(void* const* d_in, const int* in_sizes, int n_in,
                              void* d_out, int out_size, void* d_ws, size_t ws_size,
                              hipStream_t stream) {
  (void)in_sizes; (void)n_in; (void)out_size; (void)ws_size;
  const float* x     = (const float*)d_in[0];
  const float* Wq    = (const float*)d_in[1];
  const float* Wk    = (const float*)d_in[2];
  const float* Wv    = (const float*)d_in[3];
  const float* Wproj = (const float*)d_in[4];
  const float* bproj = (const float*)d_in[5];
  const float* W1    = (const float*)d_in[6];
  const float* b1    = (const float*)d_in[7];
  const float* W2    = (const float*)d_in[8];
  const float* b2    = (const float*)d_in[9];
  const float* g1    = (const float*)d_in[10];
  const float* be1   = (const float*)d_in[11];
  const float* g2    = (const float*)d_in[12];
  const float* be2   = (const float*)d_in[13];

  char* ws = (char*)d_ws;
  size_t off = 0;
  auto take = [&](size_t bytes) { char* p = ws + off; off += (bytes + 255) & ~(size_t)255; return p; };
  _Float16* qkv    = (_Float16*)take((size_t)BT * 1152 * 2);  // 75.5 MB
  _Float16* attn   = (_Float16*)take((size_t)BT * 384 * 2);   // 25.2 MB
  float*    x1     = (float*)   take((size_t)BT * 384 * 4);   // 50.3 MB
  _Float16* xn     = (_Float16*)take((size_t)BT * 384 * 2);   // 25.2 MB (LN1 out, reused for LN2 out)
  _Float16* Wqkvh  = (_Float16*)take((size_t)384 * 1152 * 2);
  _Float16* Wprojh = (_Float16*)take((size_t)384 * 384 * 2);
  _Float16* W1h    = (_Float16*)take((size_t)384 * 1536 * 2);
  _Float16* W2h    = (_Float16*)take((size_t)1536 * 384 * 2);
  _Float16* ff1    = qkv;  // reuse qkv+attn region (both dead before FF1 is written)

  // weights -> f16
  k_pack_qkv<<<(384 * 1152 + 255) / 256, 256, 0, stream>>>(Wq, Wk, Wv, Wqkvh);
  k_conv_f16<<<(384 * 384 + 255) / 256, 256, 0, stream>>>(Wproj, Wprojh, 384 * 384);
  k_conv_f16<<<(384 * 1536 + 255) / 256, 256, 0, stream>>>(W1, W1h, 384 * 1536);
  k_conv_f16<<<(1536 * 384 + 255) / 256, 256, 0, stream>>>(W2, W2h, 1536 * 384);

  // x1 = x + attn_proj(attention(LN1(x)))
  k_layernorm<<<BT, 128, 0, stream>>>(x, g1, be1, xn);
  k_gemm<0><<<dim3(1152 / 128, BT / 128), 256, 0, stream>>>(xn, Wqkvh, 384, 1152, nullptr, nullptr, qkv);
  k_attention<<<BATB * N_HEADS, 256, 0, stream>>>(qkv, attn);
  k_gemm<1><<<dim3(384 / 128, BT / 128), 256, 0, stream>>>(attn, Wprojh, 384, 384, bproj, x, x1);

  // out = x1 + W2 * relu(W1 * LN2(x1))
  k_layernorm<<<BT, 128, 0, stream>>>(x1, g2, be2, xn);
  k_gemm<2><<<dim3(1536 / 128, BT / 128), 256, 0, stream>>>(xn, W1h, 384, 1536, b1, nullptr, ff1);
  k_gemm<1><<<dim3(384 / 128, BT / 128), 256, 0, stream>>>(ff1, W2h, 1536, 384, b2, x1, d_out);
}
